// FastRCNN_7215545057805
// MI455X (gfx1250) — matrix operand order repaired and checked by reading, not compiled, not run on hardware
//
#include <hip/hip_runtime.h>
#include <hip/hip_bf16.h>
#include <stdint.h>
#include <stddef.h>

// ---------------- problem constants (match reference) ----------------
#define N_ROI   2000
#define C_CH    64
#define IMG_HW  512
#define ROI_SZ  28
#define P_GRID  7
#define BIN_SZ  4
#define K1      3136      // C*P*P = 64*49
#define DFC     4096
#define NHEAD   10        // 2 cls + 8 bbox

// ---------------- workspace layout (bytes, all 256B aligned) ---------
#define FMAP_OFF  ((size_t)0)                         // bf16 [64][512][512] = 33,554,432
#define FEAT_OFF  ((size_t)33554432)                  // bf16 [2000][3136]   = 12,544,000
#define W1_OFF    ((size_t)46098432)                  // bf16 [4096][3136]   = 25,690,112
#define W2_OFF    ((size_t)71788544)                  // bf16 [4096][4096]   = 33,554,432
#define H1_OFF    ((size_t)105342976)                 // bf16 [2000][4096]   = 16,384,000
#define H2_OFF    ((size_t)121726976)                 // bf16 [2000][4096]   = 16,384,000
#define HWT_OFF   ((size_t)138110976)                 // bf16 [10][4096]     = 81,920

// ---------------- types / helpers ------------------------------------
typedef __attribute__((ext_vector_type(16))) __bf16 v16bf;
typedef __attribute__((ext_vector_type(8)))  float  v8f;

__device__ __forceinline__ float bf2f(unsigned short u) {
  unsigned int x = ((unsigned int)u) << 16;
  return __builtin_bit_cast(float, x);
}
__device__ __forceinline__ unsigned short f2bf(float f) {
  unsigned int u = __builtin_bit_cast(unsigned int, f);
  u += 0x7FFFu + ((u >> 16) & 1u);           // round-to-nearest-even
  return (unsigned short)(u >> 16);
}

union FragU { uint4 q[2]; v16bf v; };

// ---------------- fp32 -> bf16 conversion ----------------------------
__global__ void cvt_f32_bf16(const float* __restrict__ in,
                             unsigned short* __restrict__ out, int n) {
  int i = blockIdx.x * blockDim.x + threadIdx.x;
  if (i < n) out[i] = f2bf(in[i]);
}

// ---------------- conv 3x3, pad=1, 3->64 ch, + ReLU, bf16 out --------
__global__ void conv3x3_relu(const float* __restrict__ img,
                             const float* __restrict__ cw,
                             const float* __restrict__ cb,
                             unsigned short* __restrict__ fmap) {
  int idx = blockIdx.x * blockDim.x + threadIdx.x;
  const int total = C_CH * IMG_HW * IMG_HW;
  if (idx >= total) return;
  const int x = idx & (IMG_HW - 1);
  const int y = (idx >> 9) & (IMG_HW - 1);
  const int c = idx >> 18;
  float acc = cb[c];
  const float* wc = cw + c * 27;
#pragma unroll
  for (int ic = 0; ic < 3; ++ic) {
    const float* ip = img + (size_t)ic * IMG_HW * IMG_HW;
#pragma unroll
    for (int dy = 0; dy < 3; ++dy) {
      const int yy = y + dy - 1;
      if (yy < 0 || yy >= IMG_HW) continue;
#pragma unroll
      for (int dx = 0; dx < 3; ++dx) {
        const int xx = x + dx - 1;
        if (xx < 0 || xx >= IMG_HW) continue;
        acc += ip[yy * IMG_HW + xx] * wc[ic * 9 + dy * 3 + dx];
      }
    }
  }
  acc = acc > 0.f ? acc : 0.f;
  fmap[idx] = f2bf(acc);
}

// ---------------- RoI max-pool 28x28 -> 7x7 (bins of 4x4) ------------
__global__ void roi_maxpool(const unsigned short* __restrict__ fmap,
                            const int* __restrict__ regions,
                            unsigned short* __restrict__ feat) {
  int idx = blockIdx.x * blockDim.x + threadIdx.x;
  const int total = N_ROI * C_CH * P_GRID * P_GRID;
  if (idx >= total) return;
  const int px = idx % P_GRID;  int t = idx / P_GRID;
  const int py = t % P_GRID;    t /= P_GRID;
  const int c  = t % C_CH;      const int n = t / C_CH;
  const int y0 = regions[2 * n]     + py * BIN_SZ;
  const int x0 = regions[2 * n + 1] + px * BIN_SZ;
  const unsigned short* f = fmap + (size_t)c * IMG_HW * IMG_HW;
  float m = -3.4e38f;
#pragma unroll
  for (int i = 0; i < BIN_SZ; ++i) {
    const unsigned short* row = f + (size_t)(y0 + i) * IMG_HW + x0;
#pragma unroll
    for (int j = 0; j < BIN_SZ; ++j) {
      float v = bf2f(row[j]);
      m = v > m ? v : m;
    }
  }
  feat[(size_t)n * K1 + c * (P_GRID * P_GRID) + py * P_GRID + px] = f2bf(m);
}

// ---------------- WMMA GEMM fragment helpers -------------------------
// Named scalar pointers (NOT an array) so clang's address-space inference
// keeps these as pure global_load_b128 (array alloca demoted them to flat).
struct BPtrs {
  const unsigned short* p0;
  const unsigned short* p1;
  const unsigned short* p2;
  const unsigned short* p3;
};

__device__ __forceinline__ void load_frags(const unsigned short* __restrict__ ap,
                                           const BPtrs& bp, int k,
                                           FragU& a, FragU (&b)[4]) {
  a.q[0] = *(const uint4*)(ap + k);
  a.q[1] = *(const uint4*)(ap + k + 8);
  b[0].q[0] = *(const uint4*)(bp.p0 + k);
  b[0].q[1] = *(const uint4*)(bp.p0 + k + 8);
  b[1].q[0] = *(const uint4*)(bp.p1 + k);
  b[1].q[1] = *(const uint4*)(bp.p1 + k + 8);
  b[2].q[0] = *(const uint4*)(bp.p2 + k);
  b[2].q[1] = *(const uint4*)(bp.p2 + k + 8);
  b[3].q[0] = *(const uint4*)(bp.p3 + k);
  b[3].q[1] = *(const uint4*)(bp.p3 + k + 8);
}

__device__ __forceinline__ void do_wmma(v8f (&acc)[4], FragU& a, FragU (&b)[4]) {
#pragma unroll
  for (int f = 0; f < 4; ++f)
    acc[f] = __builtin_amdgcn_wmma_f32_16x16x32_bf16(
        false, a.v, false, b[f].v, (short)0, acc[f], false, false);
}

// ---------------- WMMA GEMM: Out = relu(A @ W^T + bias) --------------
// A: [M x K] bf16 row-major; W: [N x K] bf16 row-major (so B = W^T).
// One wave computes a 16x64 output tile; 4 v_wmma per k=32 step.
// Software-pipelined: double-buffered fragments, unroll by k=64,
// sched_group_barrier enforces (10 vmem -> 4 wmma) x2 per iteration.
// Requires: K/32 even and K >= 64 (K1/32 = 98, K2/32 = 128).
__global__ void __launch_bounds__(256)
gemm_bias_relu_bf16(const unsigned short* __restrict__ A,
                    const unsigned short* __restrict__ W,
                    const float* __restrict__ bias,
                    unsigned short* __restrict__ Out,
                    int mtiles, int N, int K) {
  const int lane = threadIdx.x & 31;
  const int wave = threadIdx.x >> 5;
  const int t  = blockIdx.x * 8 + wave;
  const int mt = t % mtiles;            // m fast-varying: waves in a block share B
  const int nt = t / mtiles;
  const int m0 = mt << 4;
  const int n0 = nt << 6;

  // A fragment: lane row m0+lane%16, 16 contiguous k at k0 + 16*(lane>=16)
  const unsigned short* ap =
      A + (size_t)(m0 + (lane & 15)) * K + ((lane >> 4) << 4);
  // B fragment from W rows: lane col n, 16 contiguous k at k0 + 16*(lane>=16)
  const unsigned short* bbase =
      W + (size_t)(n0 + (lane & 15)) * K + ((lane >> 4) << 4);
  BPtrs bp;
  bp.p0 = bbase;
  bp.p1 = bbase + (size_t)16 * K;
  bp.p2 = bbase + (size_t)32 * K;
  bp.p3 = bbase + (size_t)48 * K;

  v8f acc[4] = {};
  FragU a0, a1;
  FragU b0[4], b1[4];

  // prologue: fill buffer 0 with k-step 0
  load_frags(ap, bp, 0, a0, b0);

  for (int k = 0; k < K - 64; k += 64) {
    load_frags(ap, bp, 32, a1, b1);      // next half-step, in flight over wmmas(buf0)
    do_wmma(acc, a0, b0);
    load_frags(ap, bp, 64, a0, b0);      // step k+64, in flight over wmmas(buf1)
    do_wmma(acc, a1, b1);
    ap += 64;
    bp.p0 += 64; bp.p1 += 64; bp.p2 += 64; bp.p3 += 64;
    // enforce (loads, wmmas, loads, wmmas) interleave in this region
    __builtin_amdgcn_sched_group_barrier(0x20, 10, 0);  // 10 VMEM reads
    __builtin_amdgcn_sched_group_barrier(0x8,   4, 0);  // 4 WMMA
    __builtin_amdgcn_sched_group_barrier(0x20, 10, 0);  // 10 VMEM reads
    __builtin_amdgcn_sched_group_barrier(0x8,   4, 0);  // 4 WMMA
  }
  // tail: buf0 holds k = K-64; load final half-step, no further prefetch
  load_frags(ap, bp, 32, a1, b1);
  do_wmma(acc, a0, b0);
  do_wmma(acc, a1, b1);

  // Epilogue: C layout: col = lane%16, VGPR r -> row r + 8*(lane>=16)
  const int col = lane & 15;
  const int rb  = (lane >> 4) << 3;
#pragma unroll
  for (int f = 0; f < 4; ++f) {
    const int n = n0 + (f << 4) + col;
    const float bv = bias[n];
    unsigned short* op = Out + (size_t)(m0 + rb) * N + n;
#pragma unroll
    for (int r = 0; r < 8; ++r) {
      float v = acc[f][r] + bv;
      v = v > 0.f ? v : 0.f;
      op[(size_t)r * N] = f2bf(v);
    }
  }
}

// ---------------- heads: 10 output columns, plain dot products -------
__global__ void heads_kernel(const unsigned short* __restrict__ H2,
                             const unsigned short* __restrict__ HWt,
                             const float* __restrict__ cls_b,
                             const float* __restrict__ bbox_b,
                             float* __restrict__ out) {
  int idx = blockIdx.x * blockDim.x + threadIdx.x;
  if (idx >= N_ROI * NHEAD) return;
  const int n = idx / NHEAD;
  const int j = idx - n * NHEAD;
  const unsigned short* h = H2  + (size_t)n * DFC;
  const unsigned short* w = HWt + (size_t)j * DFC;
  float s = 0.f;
  for (int k = 0; k < DFC; k += 8) {
    uint4 hq = *(const uint4*)(h + k);
    uint4 wq = *(const uint4*)(w + k);
    unsigned int hu[4] = {hq.x, hq.y, hq.z, hq.w};
    unsigned int wu[4] = {wq.x, wq.y, wq.z, wq.w};
#pragma unroll
    for (int q = 0; q < 4; ++q) {
      s += bf2f((unsigned short)(hu[q] & 0xFFFFu)) *
           bf2f((unsigned short)(wu[q] & 0xFFFFu));
      s += bf2f((unsigned short)(hu[q] >> 16)) *
           bf2f((unsigned short)(wu[q] >> 16));
    }
  }
  if (j < 2) out[n * 2 + j] = s + cls_b[j];
  else       out[N_ROI * 2 + n * 8 + (j - 2)] = s + bbox_b[j - 2];
}

// ---------------- launch ---------------------------------------------
extern "C" void kernel_launch(void* const* d_in, const int* in_sizes, int n_in,
                              void* d_out, int out_size, void* d_ws, size_t ws_size,
                              hipStream_t stream) {
  const float* img    = (const float*)d_in[0];
  const int*   regs   = (const int*)  d_in[1];
  const float* conv_w = (const float*)d_in[2];
  const float* conv_b = (const float*)d_in[3];
  const float* fc1_w  = (const float*)d_in[4];
  const float* fc1_b  = (const float*)d_in[5];
  const float* fc2_w  = (const float*)d_in[6];
  const float* fc2_b  = (const float*)d_in[7];
  const float* cls_w  = (const float*)d_in[8];
  const float* cls_b  = (const float*)d_in[9];
  const float* bbox_w = (const float*)d_in[10];
  const float* bbox_b = (const float*)d_in[11];
  float* out = (float*)d_out;

  char* ws = (char*)d_ws;
  unsigned short* fmap = (unsigned short*)(ws + FMAP_OFF);
  unsigned short* feat = (unsigned short*)(ws + FEAT_OFF);
  unsigned short* w1   = (unsigned short*)(ws + W1_OFF);
  unsigned short* w2   = (unsigned short*)(ws + W2_OFF);
  unsigned short* h1   = (unsigned short*)(ws + H1_OFF);
  unsigned short* h2   = (unsigned short*)(ws + H2_OFF);
  unsigned short* hwt  = (unsigned short*)(ws + HWT_OFF);

  const int B = 256;

  // 1) weight conversions fp32 -> bf16
  {
    int n = DFC * K1;                                       // fc1_w
    cvt_f32_bf16<<<(n + B - 1) / B, B, 0, stream>>>(fc1_w, w1, n);
  }
  {
    int n = DFC * DFC;                                      // fc2_w
    cvt_f32_bf16<<<(n + B - 1) / B, B, 0, stream>>>(fc2_w, w2, n);
  }
  {
    int n = 2 * DFC;                                        // cls_w -> hwt[0:2]
    cvt_f32_bf16<<<(n + B - 1) / B, B, 0, stream>>>(cls_w, hwt, n);
  }
  {
    int n = 8 * DFC;                                        // bbox_w -> hwt[2:10]
    cvt_f32_bf16<<<(n + B - 1) / B, B, 0, stream>>>(bbox_w, hwt + 2 * DFC, n);
  }

  // 2) conv trunk + ReLU -> bf16 fmap [64,512,512]
  {
    int n = C_CH * IMG_HW * IMG_HW;
    conv3x3_relu<<<(n + B - 1) / B, B, 0, stream>>>(img, conv_w, conv_b, fmap);
  }

  // 3) RoI max-pool -> bf16 feat [2000, 3136]
  {
    int n = N_ROI * C_CH * P_GRID * P_GRID;
    roi_maxpool<<<(n + B - 1) / B, B, 0, stream>>>(fmap, regs, feat);
  }

  // 4) FC1: h1 = relu(feat @ fc1_w^T + b). 125 m-tiles * 64 n64-tiles = 8000 waves.
  {
    const int mtiles = N_ROI / 16;          // 125
    const int ntiles = DFC / 64;            // 64
    gemm_bias_relu_bf16<<<(mtiles * ntiles) / 8, B, 0, stream>>>(
        feat, w1, fc1_b, h1, mtiles, DFC, K1);
  }

  // 5) FC2: h2 = relu(h1 @ fc2_w^T + b)
  {
    const int mtiles = N_ROI / 16;
    const int ntiles = DFC / 64;
    gemm_bias_relu_bf16<<<(mtiles * ntiles) / 8, B, 0, stream>>>(
        h1, w2, fc2_b, h2, mtiles, DFC, DFC);
  }

  // 6) heads -> d_out (cls 4000 floats, then bbox 16000 floats)
  {
    int n = N_ROI * NHEAD;
    heads_kernel<<<(n + B - 1) / B, B, 0, stream>>>(h2, hwt, cls_b, bbox_b, out);
  }
}
